// model_encdec_19885698580929
// MI455X (gfx1250) — matrix operand order repaired and checked by reading, not compiled, not run on hardware
//
#include <hip/hip_runtime.h>

// ---------------------------------------------------------------------------
// Problem constants (from reference)
// ---------------------------------------------------------------------------
#define B_N    512
#define TP     8          // T_PAST
#define MROWS  100000
#define MPAD   100352     // 196 * 512 (block tile = 512 memory rows)
#define MTILES 196
#define NCAND  200
#define NCTIL  13         // ceil(200/16) candidate tiles in K6
#define NDEC   120
#define KCL    20
#define NEG_INF (-3.402823e38f)

typedef __attribute__((ext_vector_type(16))) _Float16 v16h;
typedef __attribute__((ext_vector_type(8)))  float    v8f;

// ---------------------------------------------------------------------------
// K1: per-row encoders  flat(16) -> 64 with ReLU, plus mean-pool accumulation
// grid(B_N) block(64)
// ---------------------------------------------------------------------------
__global__ void k1_encode(const float* __restrict__ past,
                          const float* __restrict__ abs_past,
                          const float* __restrict__ W_norm, const float* __restrict__ b_norm,
                          const float* __restrict__ W_abs,  const float* __restrict__ b_abs,
                          const float* __restrict__ W_traj, const float* __restrict__ b_traj,
                          float* __restrict__ norm_state, float* __restrict__ abs_state,
                          float* __restrict__ traj_state,
                          float* __restrict__ mean_abs, float* __restrict__ mean_traj) {
  __shared__ float fp[16], fa[16];
  const int b = blockIdx.x, t = threadIdx.x;
  if (t < 16) { fp[t] = past[b * 16 + t]; fa[t] = abs_past[b * 16 + t]; }
  __syncthreads();
  float an = b_norm[t], aa = b_abs[t], at = b_traj[t];
  #pragma unroll
  for (int i = 0; i < 16; ++i) {
    an += fp[i] * W_norm[i * 64 + t];
    aa += fa[i] * W_abs[i * 64 + t];
    at += fa[i] * W_traj[i * 64 + t];
  }
  an = fmaxf(an, 0.f); aa = fmaxf(aa, 0.f); at = fmaxf(at, 0.f);
  norm_state[b * 64 + t] = an;
  abs_state[b * 64 + t]  = aa;
  traj_state[b * 64 + t] = at;
  atomicAdd(&mean_abs[t], aa);
  atomicAdd(&mean_traj[t], at);
}

// ---------------------------------------------------------------------------
// K2: social layers, state_past assembly, normalized f16 queries, qn
// grid(B_N) block(128)
// ---------------------------------------------------------------------------
__global__ void k2_social(const float* __restrict__ end_pose,
                          const float* __restrict__ norm_state,
                          const float* __restrict__ abs_state,
                          const float* __restrict__ traj_state,
                          const float* __restrict__ mean_abs,
                          const float* __restrict__ mean_traj,
                          const float* __restrict__ W_socX, const float* __restrict__ b_socX,
                          const float* __restrict__ W_int,  const float* __restrict__ b_int,
                          const float* __restrict__ W_q,    const float* __restrict__ b_q,
                          float* __restrict__ abs_soc, float* __restrict__ traj_soc,
                          float* __restrict__ state_past, _Float16* __restrict__ spn,
                          float* __restrict__ qn) {
  __shared__ float inA[130], inT[130], sp[128], red[128], ql[64];
  const int b = blockIdx.x, t = threadIdx.x;
  const float inv = 1.0f / (float)B_N;
  if (t < 64) {
    inA[t] = abs_state[b * 64 + t];  inA[64 + t] = mean_abs[t] * inv;
    inT[t] = traj_state[b * 64 + t]; inT[64 + t] = mean_traj[t] * inv;
  }
  if (t == 0) {
    inA[128] = end_pose[2 * b]; inA[129] = end_pose[2 * b + 1];
    inT[128] = end_pose[2 * b]; inT[129] = end_pose[2 * b + 1];
  }
  __syncthreads();
  if (t < 64) {
    float a = b_socX[t];
    for (int i = 0; i < 130; ++i) a += inA[i] * W_socX[i * 64 + t];
    a = fmaxf(a, 0.f);
    abs_soc[b * 64 + t] = a;
    sp[64 + t] = a;
    sp[t] = norm_state[b * 64 + t];
  } else {
    int d = t - 64;
    float a = b_int[d];
    for (int i = 0; i < 130; ++i) a += inT[i] * W_int[i * 64 + d];
    traj_soc[b * 64 + d] = fmaxf(a, 0.f);
  }
  __syncthreads();
  red[t] = sp[t] * sp[t];
  __syncthreads();
  for (int s = 64; s > 0; s >>= 1) { if (t < s) red[t] += red[t + s]; __syncthreads(); }
  const float scale = rsqrtf(fmaxf(red[0], 1e-24f));
  __syncthreads();
  state_past[b * 128 + t] = sp[t];
  spn[(size_t)b * 128 + t] = (_Float16)(sp[t] * scale);
  // query = state_past @ W_query + b, then normalize
  if (t < 64) {
    float q = b_q[t];
    for (int i = 0; i < 128; ++i) q += sp[i] * W_q[i * 64 + t];
    ql[t] = q; red[t] = q * q;
  }
  __syncthreads();
  for (int s = 32; s > 0; s >>= 1) { if (t < s) red[t] += red[t + s]; __syncthreads(); }
  if (t < 64) qn[b * 64 + t] = ql[t] * rsqrtf(fmaxf(red[0], 1e-24f));
}

// ---------------------------------------------------------------------------
// K3: normalize memory rows -> f16 (pad rows zero)   grid(MPAD) block(128)
// ---------------------------------------------------------------------------
__global__ void k3_memnorm(const float* __restrict__ memory_past,
                           _Float16* __restrict__ mp16) {
  __shared__ float red[128];
  const int r = blockIdx.x, t = threadIdx.x;
  if (r >= MROWS) { mp16[(size_t)r * 128 + t] = (_Float16)0.f; return; }
  const float v = memory_past[(size_t)r * 128 + t];
  red[t] = v * v;
  __syncthreads();
  for (int s = 64; s > 0; s >>= 1) { if (t < s) red[t] += red[t + s]; __syncthreads(); }
  mp16[(size_t)r * 128 + t] = (_Float16)(v * rsqrtf(fmaxf(red[0], 1e-24f)));
}

// ---------------------------------------------------------------------------
// K4: big cosine-similarity GEMM via WMMA f16 -> f32
// scores[512][MPAD] = spn(512x128) @ mp16^T(128xMPAD)
// grid(MTILES, 32) block(256): 8 waves; each wave computes a 16(query)x64(mem)
// register-blocked tile (4 accumulators, A chunk reused across 4 B tiles).
// ---------------------------------------------------------------------------
__global__ void k4_wmma_scores(const _Float16* __restrict__ spn,
                               const _Float16* __restrict__ mp16,
                               float* __restrict__ scores) {
  const int lane = threadIdx.x & 31;
  const int wv   = threadIdx.x >> 5;
  const int l15  = lane & 15;
  const int hi   = lane >> 4;
  const int qBase = blockIdx.y * 16;
  const size_t mWave = (size_t)blockIdx.x * 512 + (size_t)wv * 64;

  const _Float16* arow = spn + (size_t)(qBase + l15) * 128 + hi * 16;

  if (blockIdx.x + 1 < MTILES)
    __builtin_prefetch(mp16 + (mWave + 512 + l15) * 128, 0, 1);

  const v8f z = {};
  v8f acc[4] = {z, z, z, z};
  #pragma unroll
  for (int kc = 0; kc < 4; ++kc) {
    v16h a;
    // A: 16x32 f16, this code's contiguous K order (B matches) — lanes 0-15: M=l15, K in {0..15}; lanes 16-31: K in {16..31}
    #pragma unroll
    for (int i = 0; i < 8; ++i) {
      a[i]     = arow[kc * 32 + i];
      a[i + 8] = arow[kc * 32 + 8 + i];
    }
    #pragma unroll
    for (int n = 0; n < 4; ++n) {
      // B: 32x16 f16 — lane holds column N=l15, contiguous K range per half-wave
      const _Float16* brow = mp16 + (mWave + n * 16 + l15) * 128 + kc * 32 + hi * 16;
      v16h bm;
      #pragma unroll
      for (int i = 0; i < 16; ++i) bm[i] = brow[i];
      acc[n] = __builtin_amdgcn_wmma_f32_16x16x32_f16(false, a, false, bm,
                                                      (short)0, acc[n], false, false);
    }
  }
  // C/D layout: VGPR r, lanes 0-15 -> M=r, lanes 16-31 -> M=r+8; N=l15
  #pragma unroll
  for (int n = 0; n < 4; ++n) {
    const size_t col = mWave + n * 16 + l15;
    #pragma unroll
    for (int r = 0; r < 8; ++r) {
      scores[(size_t)(qBase + r + hi * 8) * MPAD + col] = acc[n][r];
    }
  }
}

// ---------------------------------------------------------------------------
// K5: top-200 per row (descending, lowest-index tiebreak like lax.top_k)
// grid(B_N) block(256); destructively masks scores (K4 rewrites every call)
// ---------------------------------------------------------------------------
__global__ void k5_top200(float* __restrict__ scores, int* __restrict__ idx200) {
  __shared__ float sv[256];
  __shared__ int   si[256];
  const int b = blockIdx.x, t = threadIdx.x;
  float* row = scores + (size_t)b * MPAD;
  for (int j = 0; j < NCAND; ++j) {
    float best = NEG_INF; int bi = 0x7fffffff;
    for (int m = t; m < MROWS; m += 256) {
      const float v = row[m];
      if (v > best) { best = v; bi = m; }
    }
    sv[t] = best; si[t] = bi;
    __syncthreads();
    for (int s = 128; s > 0; s >>= 1) {
      if (t < s) {
        const float ov = sv[t + s]; const int oi = si[t + s];
        if (ov > sv[t] || (ov == sv[t] && oi < si[t])) { sv[t] = ov; si[t] = oi; }
      }
      __syncthreads();
    }
    if (t == 0) { idx200[b * NCAND + j] = si[0]; row[si[0]] = NEG_INF; }
    __syncthreads();
  }
}

// ---------------------------------------------------------------------------
// K6: attention re-rank via WMMA + top-120
// grid(B_N) block(256).
//   wT  : W_mem transposed to f16 in LDS, [n][k] contiguous-K (B operand)
//   a16 : 16 gathered candidate rows converted to f16 (A operand)
//   per candidate tile: waves 0-3 each produce 16cand x 16dim via 4 WMMAs,
//   result + b_mem staged to LDS, cosine vs qn computed, then block top-120.
// idx120 stores GLOBAL memory row ids.
// ---------------------------------------------------------------------------
__global__ void k6_attn_top120(const float* __restrict__ memory_past,
                               const float* __restrict__ W_mem, const float* __restrict__ b_mem,
                               const float* __restrict__ qn,
                               const int* __restrict__ idx200,
                               int* __restrict__ idx120) {
  __shared__ _Float16 wT[64 * 128];     // B: [n][k]
  __shared__ _Float16 a16[16 * 128];    // A: [cand][k]
  __shared__ float mkst[16 * 64];       // mk tile (cand x dim)
  __shared__ float qnl[64];
  __shared__ float w2l[NCTIL * 16];     // 208, only first 200 ranked
  __shared__ float rv[256];
  __shared__ int   ri[256];
  const int b = blockIdx.x, t = threadIdx.x;
  const int lane = t & 31, wv = t >> 5, l15 = lane & 15, hi = lane >> 4;

  // transpose-convert W_mem (128x64 f32) -> wT (64x128 f16)
  for (int idx = t; idx < 64 * 128; idx += 256) {
    const int n = idx >> 7, k = idx & 127;
    wT[n * 128 + k] = (_Float16)W_mem[k * 64 + n];
  }
  if (t < 64) qnl[t] = qn[b * 64 + t];
  const float bias = (wv < 4) ? b_mem[wv * 16 + l15] : 0.f;
  __syncthreads();

  for (int jt = 0; jt < NCTIL; ++jt) {
    // gather + convert 16 candidate rows (clamp past 200 -> duplicates, ignored)
    {
      const int r = t >> 4;                     // 0..15
      const int c = (t & 15) * 8;               // 0..120 step 8
      int j = jt * 16 + r; if (j > NCAND - 1) j = NCAND - 1;
      const float* mrow = memory_past + (size_t)idx200[b * NCAND + j] * 128;
      #pragma unroll
      for (int i = 0; i < 8; ++i) a16[r * 128 + c + i] = (_Float16)mrow[c + i];
    }
    __syncthreads();

    if (wv < 4) {
      const v8f zz = {};
      v8f acc = zz;
      #pragma unroll
      for (int kc = 0; kc < 4; ++kc) {
        v16h a, bm;
        const _Float16* ar = &a16[l15 * 128 + kc * 32 + hi * 16];
        #pragma unroll
        for (int i = 0; i < 8; ++i) { a[i] = ar[i]; a[i + 8] = ar[8 + i]; }
        const _Float16* br = &wT[(wv * 16 + l15) * 128 + kc * 32 + hi * 16];
        #pragma unroll
        for (int i = 0; i < 16; ++i) bm[i] = br[i];
        acc = __builtin_amdgcn_wmma_f32_16x16x32_f16(false, a, false, bm,
                                                     (short)0, acc, false, false);
      }
      #pragma unroll
      for (int r = 0; r < 8; ++r)
        mkst[(r + hi * 8) * 64 + wv * 16 + l15] = acc[r] + bias;
    }
    __syncthreads();

    if (t < 16) {
      float ss = 0.f, dot = 0.f;
      for (int d = 0; d < 64; ++d) {
        const float v = mkst[t * 64 + d];
        ss += v * v; dot += v * qnl[d];
      }
      w2l[jt * 16 + t] = dot * rsqrtf(fmaxf(ss, 1e-24f));
    }
    __syncthreads();
  }

  // top-120 of the 200 (descending, lowest-index tiebreak)
  for (int s = 0; s < NDEC; ++s) {
    float best = NEG_INF; int bi = 0x7fffffff;
    if (t < NCAND) { best = w2l[t]; bi = t; }
    rv[t] = best; ri[t] = bi;
    __syncthreads();
    for (int r = 128; r > 0; r >>= 1) {
      if (t < r) {
        const float ov = rv[t + r]; const int oi = ri[t + r];
        if (ov > rv[t] || (ov == rv[t] && oi < ri[t])) { rv[t] = ov; ri[t] = oi; }
      }
      __syncthreads();
    }
    if (t == 0) {
      idx120[b * NDEC + s] = idx200[b * NCAND + ri[0]];
      w2l[ri[0]] = NEG_INF;
    }
    __syncthreads();
  }
}

// ---------------------------------------------------------------------------
// K7: candidate decoder (y1 + residual-reconstruction y2) -> cand(b,120,2)
// grid(B_N) block(64)
// ---------------------------------------------------------------------------
__global__ void k7_cand(const float* __restrict__ past,
                        const float* __restrict__ memory_fut,
                        const float* __restrict__ state_past,
                        const float* __restrict__ abs_soc,
                        const int* __restrict__ idx120,
                        const float* __restrict__ W_dec,  const float* __restrict__ b_dec,
                        const float* __restrict__ W_decx, const float* __restrict__ b_decx,
                        const float* __restrict__ W_res,  const float* __restrict__ b_res,
                        const float* __restrict__ W_dec2, const float* __restrict__ b_dec2,
                        float* __restrict__ cand) {
  __shared__ float spl[128], socl[64], pastl[16], ffl[64], diffl[16], dembl[64], y1l[2];
  const int b = blockIdx.x, t = threadIdx.x;
  spl[t]      = state_past[b * 128 + t];
  spl[64 + t] = state_past[b * 128 + 64 + t];
  socl[t]     = abs_soc[b * 64 + t];
  if (t < 16) pastl[t] = past[b * 16 + t];
  __syncthreads();
  for (int j = 0; j < NDEC; ++j) {
    const int row = idx120[b * NDEC + j];
    ffl[t] = memory_fut[(size_t)row * 64 + t];
    __syncthreads();
    if (t < 2) {
      float a = b_dec[t];
      for (int i = 0; i < 192; ++i) {
        const float sc = (i < 128) ? spl[i] : ffl[i - 128];
        a += sc * W_dec[i * 2 + t];
      }
      y1l[t] = a;
    }
    if (t < 16) {
      float r = b_decx[t];
      for (int i = 0; i < 192; ++i) {
        const float sc = (i < 128) ? spl[i] : ffl[i - 128];
        r += sc * W_decx[i * 16 + t];
      }
      diffl[t] = pastl[t] - r;
    }
    __syncthreads();
    float de = b_res[t];
    #pragma unroll
    for (int i = 0; i < 16; ++i) de += diffl[i] * W_res[i * 64 + t];
    dembl[t] = fmaxf(de, 0.f);
    __syncthreads();
    if (t < 2) {
      float a = b_dec2[t];
      for (int i = 0; i < 192; ++i) {
        const float sc = (i < 64) ? dembl[i] : ((i < 128) ? socl[i - 64] : ffl[i - 128]);
        a += sc * W_dec2[i * 2 + t];
      }
      cand[((size_t)b * NDEC + j) * 2 + t] = y1l[t] + a;
    }
    __syncthreads();
  }
}

// ---------------------------------------------------------------------------
// K8: deterministic k-means (20 clusters, 10 iters) on cand(b,120,2)
// grid(B_N) block(128).  LCG Fisher-Yates init stands in for JAX threefry.
// ---------------------------------------------------------------------------
__device__ __forceinline__ unsigned lcg_next(unsigned& s) {
  s = s * 1664525u + 1013904223u;
  return s >> 8;
}

__global__ void k8_kmeans(const float* __restrict__ cand, float* __restrict__ dest) {
  __shared__ float px[NDEC], py[NDEC], cx[KCL], cy[KCL], sx[KCL], sy[KCL];
  __shared__ int   cn[KCL], perm[NDEC];
  const int b = blockIdx.x, t = threadIdx.x;
  if (t < NDEC) {
    px[t] = cand[((size_t)b * NDEC + t) * 2 + 0];
    py[t] = cand[((size_t)b * NDEC + t) * 2 + 1];
  }
  if (t == 0) {
    unsigned seed = (unsigned)b * 2654435761u + 12345u;
    for (int i = 0; i < NDEC; ++i) perm[i] = i;
    for (int i = NDEC - 1; i > 0; --i) {
      const int r = (int)(lcg_next(seed) % (unsigned)(i + 1));
      const int tmp = perm[i]; perm[i] = perm[r]; perm[r] = tmp;
    }
  }
  __syncthreads();
  if (t < KCL) { cx[t] = px[perm[t]]; cy[t] = py[perm[t]]; }
  __syncthreads();
  for (int it = 0; it < 10; ++it) {
    if (t < KCL) { sx[t] = 0.f; sy[t] = 0.f; cn[t] = 0; }
    __syncthreads();
    if (t < NDEC) {
      float bd = 3.4e38f; int bk = 0;
      for (int k = 0; k < KCL; ++k) {
        const float dx = px[t] - cx[k], dy = py[t] - cy[k];
        const float d = dx * dx + dy * dy;
        if (d < bd) { bd = d; bk = k; }
      }
      atomicAdd(&sx[bk], px[t]);
      atomicAdd(&sy[bk], py[t]);
      atomicAdd(&cn[bk], 1);
    }
    __syncthreads();
    if (t < KCL) {
      if (cn[t] > 0) {
        cx[t] = sx[t] / (float)cn[t];
        cy[t] = sy[t] / (float)cn[t];
      } else {
        unsigned h = ((unsigned)b * 131071u + (unsigned)it * 8191u + (unsigned)t * 131u + 7u) * 2654435761u;
        const int idx = (int)((h >> 7) % NDEC);
        cx[t] = px[idx]; cy[t] = py[idx];
      }
    }
    __syncthreads();
  }
  if (t < KCL) {
    dest[((size_t)b * KCL + t) * 2 + 0] = cx[t];
    dest[((size_t)b * KCL + t) * 2 + 1] = cy[t];
  }
}

// ---------------------------------------------------------------------------
// K9: destination-conditioned 2-iteration trajectory decoder -> out(512,20,12,2)
// grid(B_N) block(64)
// ---------------------------------------------------------------------------
__global__ void k9_final(const float* __restrict__ past,
                         const float* __restrict__ traj_soc,
                         const float* __restrict__ dest,
                         const float* __restrict__ W_dest, const float* __restrict__ b_dest,
                         const float* __restrict__ W_dx,   const float* __restrict__ b_dx,
                         const float* __restrict__ W_dy,   const float* __restrict__ b_dy,
                         float* __restrict__ out) {
  __shared__ float ts[64], pastl[16], inp[144], predl[22], xh[16];
  const int b = blockIdx.x, t = threadIdx.x;
  ts[t] = traj_soc[b * 64 + t];
  if (t < 16) pastl[t] = past[b * 16 + t];
  __syncthreads();
  for (int k = 0; k < KCL; ++k) {
    const float dxv = dest[((size_t)b * KCL + k) * 2 + 0];
    const float dyv = dest[((size_t)b * KCL + k) * 2 + 1];
    // dest_feat = relu(dest @ W_dest + b_dest); stc = [traj_soc, dest_feat]
    inp[80 + t] = fmaxf(b_dest[t] + dxv * W_dest[t] + dyv * W_dest[64 + t], 0.f);
    inp[16 + t] = ts[t];
    if (t < 16) inp[t] = pastl[t];       // x_hat = 0 initially
    if (t < 22) predl[t] = 0.f;
    __syncthreads();
    for (int i = 0; i < 2; ++i) {
      if (t < 16) {
        float xv = b_dx[i * 16 + t];
        for (int r = 0; r < 144; ++r) xv += inp[r] * W_dx[(i * 144 + r) * 16 + t];
        xh[t] = tanhf(xv);
      }
      if (t < 22) {
        float pv = b_dy[i * 22 + t];
        for (int r = 0; r < 144; ++r) pv += inp[r] * W_dy[(i * 144 + r) * 22 + t];
        predl[t] += pv;
      }
      __syncthreads();
      if (t < 16) inp[t] = pastl[t] - xh[t];
      __syncthreads();
    }
    float* ob = out + (((size_t)b * KCL + k) * 12) * 2;
    if (t < 11) {
      const float f = (float)(t + 1) / 12.0f;
      ob[t * 2 + 0] = predl[t * 2 + 0] + dxv * f;
      ob[t * 2 + 1] = predl[t * 2 + 1] + dyv * f;
    } else if (t == 11) {
      ob[11 * 2 + 0] = dxv;
      ob[11 * 2 + 1] = dyv;
    }
    __syncthreads();
  }
}

// ---------------------------------------------------------------------------
// host launcher
// ---------------------------------------------------------------------------
extern "C" void kernel_launch(void* const* d_in, const int* in_sizes, int n_in,
                              void* d_out, int out_size, void* d_ws, size_t ws_size,
                              hipStream_t stream) {
  (void)in_sizes; (void)n_in; (void)out_size; (void)ws_size;
  const float* past        = (const float*)d_in[0];
  const float* abs_past    = (const float*)d_in[1];
  const float* end_pose    = (const float*)d_in[2];
  const float* memory_past = (const float*)d_in[3];
  const float* memory_fut  = (const float*)d_in[4];
  // params in _params() insertion order
  const float* W_norm_past = (const float*)d_in[5];
  const float* b_norm_past = (const float*)d_in[6];
  const float* W_abs_past  = (const float*)d_in[7];
  const float* b_abs_past  = (const float*)d_in[8];
  const float* W_traj_abs  = (const float*)d_in[9];
  const float* b_traj_abs  = (const float*)d_in[10];
  const float* W_social_X  = (const float*)d_in[11];
  const float* b_social_X  = (const float*)d_in[12];
  const float* W_inter     = (const float*)d_in[13];
  const float* b_inter     = (const float*)d_in[14];
  const float* W_query     = (const float*)d_in[15];
  const float* b_query     = (const float*)d_in[16];
  const float* W_mem       = (const float*)d_in[17];
  const float* b_mem       = (const float*)d_in[18];
  const float* W_dec       = (const float*)d_in[19];
  const float* b_dec       = (const float*)d_in[20];
  const float* W_dec_x     = (const float*)d_in[21];
  const float* b_dec_x     = (const float*)d_in[22];
  const float* W_res       = (const float*)d_in[23];
  const float* b_res       = (const float*)d_in[24];
  const float* W_dec2      = (const float*)d_in[25];
  const float* b_dec2      = (const float*)d_in[26];
  const float* W_dest      = (const float*)d_in[27];
  const float* b_dest      = (const float*)d_in[28];
  const float* W_dx        = (const float*)d_in[29];
  const float* b_dx        = (const float*)d_in[30];
  const float* W_dy        = (const float*)d_in[31];
  const float* b_dy        = (const float*)d_in[32];
  // d_in[33] = seq_start_end ([[0, B]] -> full-batch pooling, handled implicitly)

  // workspace carve-up (256B-aligned regions)
  char* w = (char*)d_ws;
  auto take = [&](size_t bytes) {
    char* p = w;
    w += (bytes + 255) & ~(size_t)255;
    return p;
  };
  float*     ws_state   = (float*)take((size_t)B_N * 128 * 4);
  _Float16*  ws_spn     = (_Float16*)take((size_t)B_N * 128 * 2);
  float*     ws_qn      = (float*)take((size_t)B_N * 64 * 4);
  float*     ws_abs_soc = (float*)take((size_t)B_N * 64 * 4);
  float*     ws_trj_soc = (float*)take((size_t)B_N * 64 * 4);
  float*     ws_norm_st = (float*)take((size_t)B_N * 64 * 4);
  float*     ws_abs_st  = (float*)take((size_t)B_N * 64 * 4);
  float*     ws_traj_st = (float*)take((size_t)B_N * 64 * 4);
  float*     ws_mean_a  = (float*)take(64 * 4);
  float*     ws_mean_t  = (float*)take(64 * 4);
  _Float16*  ws_mp16    = (_Float16*)take((size_t)MPAD * 128 * 2);
  int*       ws_idx200  = (int*)take((size_t)B_N * NCAND * 4);
  int*       ws_idx120  = (int*)take((size_t)B_N * NDEC * 4);
  float*     ws_cand    = (float*)take((size_t)B_N * NDEC * 2 * 4);
  float*     ws_dest    = (float*)take((size_t)B_N * KCL * 2 * 4);
  float*     ws_scores  = (float*)take((size_t)B_N * MPAD * 4);   // ~206 MB

  hipMemsetAsync(ws_mean_a, 0, 64 * 4, stream);
  hipMemsetAsync(ws_mean_t, 0, 64 * 4, stream);

  k1_encode<<<B_N, 64, 0, stream>>>(past, abs_past,
      W_norm_past, b_norm_past, W_abs_past, b_abs_past, W_traj_abs, b_traj_abs,
      ws_norm_st, ws_abs_st, ws_traj_st, ws_mean_a, ws_mean_t);

  k2_social<<<B_N, 128, 0, stream>>>(end_pose, ws_norm_st, ws_abs_st, ws_traj_st,
      ws_mean_a, ws_mean_t, W_social_X, b_social_X, W_inter, b_inter,
      W_query, b_query, ws_abs_soc, ws_trj_soc, ws_state, ws_spn, ws_qn);

  k3_memnorm<<<MPAD, 128, 0, stream>>>(memory_past, ws_mp16);

  k4_wmma_scores<<<dim3(MTILES, B_N / 16), 256, 0, stream>>>(ws_spn, ws_mp16, ws_scores);

  k5_top200<<<B_N, 256, 0, stream>>>(ws_scores, ws_idx200);

  k6_attn_top120<<<B_N, 256, 0, stream>>>(memory_past, W_mem, b_mem, ws_qn,
                                          ws_idx200, ws_idx120);

  k7_cand<<<B_N, 64, 0, stream>>>(past, memory_fut, ws_state, ws_abs_soc, ws_idx120,
      W_dec, b_dec, W_dec_x, b_dec_x, W_res, b_res, W_dec2, b_dec2, ws_cand);

  k8_kmeans<<<B_N, 128, 0, stream>>>(ws_cand, ws_dest);

  k9_final<<<B_N, 64, 0, stream>>>(past, ws_trj_soc, ws_dest,
      W_dest, b_dest, W_dx, b_dx, W_dy, b_dy, (float*)d_out);
}
